// MobileCapsule_25915832664354
// MI455X (gfx1250) — matrix operand order repaired and checked by reading, not compiled, not run on hardware
//
#include <hip/hip_runtime.h>
#include <math.h>

// MobileCapsule forward for MI455X (gfx1250, wave32).
// - Stage GEMM (primary pose, 28800x64x128) uses v_wmma_f32_16x16x32_f16.
// - Depthwise caps: EM with Cout=1 collapses to one M-step (softmax over a
//   singleton axis is identity) -> single weighted mean/var pass.
// - ConvCaps (Cout=16): lanes<->couts, votes staged in LDS, 3-iter EM with
//   16-lane butterfly softmax.
// - ClassCaps (Bin=1024,E=10): waves<->couts, r matrix in 40KB dynamic LDS.
// Whole ~40MB intermediate set fits the 192MB L2 -> latency bound; one fused
// kernel per stage.

#define EPSV        1e-8f
#define LAMBDAV     1e-3f
#define LN2PI_HALFV 0.91893853320467274178f

typedef __attribute__((ext_vector_type(16))) _Float16 v16h;
typedef __attribute__((ext_vector_type(8)))  float    v8f;

__device__ __forceinline__ float sigmoidf_(float x) { return 1.f / (1.f + __expf(-x)); }

__device__ __forceinline__ float grp16_sum(float v) {
#pragma unroll
  for (int m = 1; m < 16; m <<= 1) v += __shfl_xor(v, m, 32);
  return v;
}
__device__ __forceinline__ float grp16_max(float v) {
#pragma unroll
  for (int m = 1; m < 16; m <<= 1) v = fmaxf(v, __shfl_xor(v, m, 32));
  return v;
}
__device__ __forceinline__ float wave_sum32(float v) {
#pragma unroll
  for (int m = 1; m < 32; m <<= 1) v += __shfl_xor(v, m, 32);
  return v;
}

// ---------------- Stage 1: conv 5x5 s2 VALID + bias + relu ------------------
// x:(32,64,64,1)  w:(5,5,1,64)  y:(32,30,30,64)
__global__ void conv1_kernel(const float* __restrict__ x, const float* __restrict__ w,
                             const float* __restrict__ b, float* __restrict__ y) {
  int t = blockIdx.x * blockDim.x + threadIdx.x;
  if (t >= 32 * 30 * 30 * 64) return;
  int c = t & 63;
  int m = t >> 6;
  int ow = m % 30; int tt = m / 30;
  int oh = tt % 30; int n = tt / 30;
  float s = b[c];
#pragma unroll
  for (int ki = 0; ki < 5; ++ki)
#pragma unroll
    for (int kj = 0; kj < 5; ++kj)
      s += x[((size_t)n * 64 + (oh * 2 + ki)) * 64 + (ow * 2 + kj)] * w[(ki * 5 + kj) * 64 + c];
  y[t] = fmaxf(s, 0.f);
}

// ---------------- Stage 2: primary pose GEMM via WMMA -----------------------
// Y:(28800,64) @ W:(64,128) + bias -> pose:(28800,128).  One wave per 16x16 tile.
__global__ void prim_pose_wmma(const float* __restrict__ Y, const float* __restrict__ W,
                               const float* __restrict__ bias, float* __restrict__ out) {
  __shared__ _Float16 sA[16 * 64];
  __shared__ _Float16 sB[64 * 16];
  const int nTiles = 128 / 16;                 // 8 N-tiles
  int m0 = (blockIdx.x / nTiles) * 16;
  int n0 = (blockIdx.x % nTiles) * 16;
  int lane = threadIdx.x;                      // 0..31

  for (int i = lane; i < 16 * 64; i += 32) {   // A tile 16xK(=64)
    int r = i >> 6, c = i & 63;
    sA[i] = (_Float16)Y[(size_t)(m0 + r) * 64 + c];
  }
  for (int i = lane; i < 64 * 16; i += 32) {   // B tile Kx16
    int r = i >> 4, c = i & 15;
    sB[i] = (_Float16)W[(size_t)r * 128 + (n0 + c)];
  }
  __syncthreads();

  v8f acc = {};
  int mA = lane & 15;
  int kh = (lane >> 4) * 8;    // A: lanes 0-15 hold K 0..7 & 16..23; 16-31 hold 8..15 & 24..31
  int nB = lane & 15;
  int kb = (lane >> 4) * 8;    // B: lanes 0-15 hold K 0..7 & 16..23; 16-31 hold 8..15 & 24..31
#pragma unroll
  for (int kk = 0; kk < 64; kk += 32) {
    v16h a, bf;
#pragma unroll
    for (int j = 0; j < 8; ++j) {
      a[j]     = sA[mA * 64 + kk + kh + j];
      a[j + 8] = sA[mA * 64 + kk + 16 + kh + j];
    }
#pragma unroll
    for (int j = 0; j < 8; ++j) {
      bf[j]     = sB[(kk + kb + j) * 16 + nB];
      bf[j + 8] = sB[(kk + 16 + kb + j) * 16 + nB];
    }
    acc = __builtin_amdgcn_wmma_f32_16x16x32_f16(false, a, false, bf, (short)0, acc, false, false);
  }
  int mh = (lane >> 4) * 8;                    // C/D: vgpr r -> M = r + 8*(lane>=16)
#pragma unroll
  for (int r = 0; r < 8; ++r)
    out[(size_t)(m0 + mh + r) * 128 + n0 + nB] = acc[r] + bias[n0 + nB];
}

// ---------------- Stage 2b: primary activation ------------------------------
// a = sigmoid(Y @ Wa(64,8) + b) -> (28800,8)
__global__ void prim_a_kernel(const float* __restrict__ Y, const float* __restrict__ W,
                              const float* __restrict__ b, float* __restrict__ out) {
  int t = blockIdx.x * blockDim.x + threadIdx.x;
  if (t >= 28800 * 8) return;
  int m = t >> 3, c = t & 7;
  float s = b[c];
#pragma unroll 8
  for (int k = 0; k < 64; ++k) s += Y[(size_t)m * 64 + k] * W[k * 8 + c];
  out[t] = sigmoidf_(s);
}

// ---------------- Depthwise caps (Cout=1 -> EM collapses to one M-step) -----
// 16 threads per (pos,capsule) problem; component = tid&15.
template <int K>
__global__ void dw_caps_kernel(const float* __restrict__ pose_in, const float* __restrict__ a_in,
                               const float* __restrict__ Wt, const float* __restrict__ bu,
                               const float* __restrict__ ba,
                               float* __restrict__ pose_out, float* __restrict__ a_out,
                               int B, int H, int Wd, int Bc, int OH, int OW) {
  const int KK = K * K, PAD = K / 2;
  int gtid = blockIdx.x * blockDim.x + threadIdx.x;
  int comp = gtid & 15;
  int prob = gtid >> 4;
  int nProb = B * OH * OW * Bc;
  if (prob >= nProb) return;
  int c = prob % Bc;
  int t = prob / Bc;
  int ow = t % OW; t /= OW;
  int oh = t % OH; int n = t / OH;
  int i = comp >> 2, l = comp & 3;

  float votes[KK], acts[KK];
  float rsum = 0.f;
#pragma unroll
  for (int k = 0; k < KK; ++k) {
    int ki = k / K, kj = k % K;
    int ih = oh * 2 + ki - PAD;
    int iw = ow * 2 + kj - PAD;
    float a = 0.f, v = 0.f;
    if (ih >= 0 && ih < H && iw >= 0 && iw < Wd) {
      size_t base = (size_t)(n * H + ih) * Wd + iw;
      a = a_in[base * Bc + c];
      const float* p = pose_in + base * (size_t)Bc * 16 + c * 16 + i * 4;
      const float* w = Wt + ((size_t)k * Bc + c) * 16 + l;
      v = p[0] * w[0] + p[1] * w[4] + p[2] * w[8] + p[3] * w[12];
    }
    votes[k] = v; acts[k] = a; rsum += a;
  }
  float inv = 1.f / (rsum + EPSV);
  float mu = 0.f;
#pragma unroll
  for (int k = 0; k < KK; ++k) mu += acts[k] * votes[k];
  mu *= inv;
  float sg = 0.f;
#pragma unroll
  for (int k = 0; k < KK; ++k) { float d = votes[k] - mu; sg += acts[k] * d * d; }
  sg = sg * inv + EPSV;
  float cost = (bu[0] + 0.5f * __logf(sg)) * rsum;
  float cs = grp16_sum(cost);
  pose_out[(size_t)prob * 16 + comp] = mu;
  if (comp == 0) a_out[prob] = sigmoidf_(LAMBDAV * (ba[0] - cs));
}

// ---------------- ConvCaps 1x1 (Cout=16), full 3-iter EM --------------------
// 2 problems per 64-thread block; one wave per problem; lane&15 <-> cout.
template <int BIN>
__global__ void cc_caps_kernel(const float* __restrict__ pose_in, const float* __restrict__ a_in,
                               const float* __restrict__ Wt, const float* __restrict__ bu,
                               const float* __restrict__ ba,
                               float* __restrict__ pose_out, float* __restrict__ a_out, int nProb) {
  __shared__ float V[2 * BIN * 256];
  __shared__ float P[2 * BIN * 16];
  int wave = threadIdx.x >> 5;
  int lane = threadIdx.x & 31;
  int prob = blockIdx.x * 2 + wave;
  bool valid = prob < nProb;
  float* Vw = V + (size_t)wave * BIN * 256;
  float* Pw = P + (size_t)wave * BIN * 16;

  if (valid)
    for (int idx = lane; idx < BIN * 16; idx += 32)
      Pw[idx] = pose_in[(size_t)prob * BIN * 16 + idx];
  __syncthreads();
  if (valid)
    for (int idx = lane; idx < BIN * 256; idx += 32) {
      int bin = idx >> 8, rem = idx & 255, co = rem >> 4, comp = rem & 15;
      int i = comp >> 2, l = comp & 3;
      const float* p = Pw + bin * 16 + i * 4;
      const float* w = Wt + ((size_t)bin * 16 + co) * 16 + l;
      Vw[idx] = p[0] * w[0] + p[1] * w[4] + p[2] * w[8] + p[3] * w[12];
    }
  __syncthreads();
  if (!valid) return;  // uniform per wave; both waves already passed barriers

  int co = lane & 15;
  float ain[BIN], r[BIN], mu[16], sg[16];
  float aout = 0.f;
#pragma unroll
  for (int bn = 0; bn < BIN; ++bn) {
    ain[bn] = a_in[(size_t)prob * BIN + bn];
    r[bn] = 1.f / 16.f;
  }
  for (int it = 0; it < 3; ++it) {
    float rw[BIN], rsum = 0.f;
#pragma unroll
    for (int bn = 0; bn < BIN; ++bn) { rw[bn] = r[bn] * ain[bn]; rsum += rw[bn]; }
    float inv = 1.f / (rsum + EPSV);
#pragma unroll
    for (int comp = 0; comp < 16; ++comp) {
      float m = 0.f;
#pragma unroll
      for (int bn = 0; bn < BIN; ++bn) m += rw[bn] * Vw[bn * 256 + co * 16 + comp];
      mu[comp] = m * inv;
    }
#pragma unroll
    for (int comp = 0; comp < 16; ++comp) {
      float s = 0.f;
#pragma unroll
      for (int bn = 0; bn < BIN; ++bn) {
        float d = Vw[bn * 256 + co * 16 + comp] - mu[comp];
        s += rw[bn] * d * d;
      }
      sg[comp] = s * inv + EPSV;
    }
    float cost = 0.f;
#pragma unroll
    for (int comp = 0; comp < 16; ++comp) cost += bu[co] + 0.5f * __logf(sg[comp]);
    cost *= rsum;
    aout = sigmoidf_(LAMBDAV * (ba[co] - cost));
    if (it < 2) {
      float lnA = __logf(EPSV + aout);
#pragma unroll
      for (int bn = 0; bn < BIN; ++bn) {
        float lp = 0.f;
#pragma unroll
        for (int comp = 0; comp < 16; ++comp) {
          float d = Vw[bn * 256 + co * 16 + comp] - mu[comp];
          lp += -d * d / (2.f * sg[comp]) - 0.5f * __logf(sg[comp]) - LN2PI_HALFV;
        }
        float la = lp + lnA;
        float mx = grp16_max(la);
        float e = __expf(la - mx);
        float se = grp16_sum(e);
        r[bn] = e / se;
      }
    }
  }
  if (lane < 16) {
#pragma unroll
    for (int comp = 0; comp < 16; ++comp)
      pose_out[((size_t)prob * 16 + co) * 16 + comp] = mu[comp];
    a_out[(size_t)prob * 16 + co] = aout;
  }
}

// ---------------- Class caps: Bin=1024 (8x8x16), E=10, coord-add ------------
__device__ __forceinline__ void cls_vote(const float* __restrict__ pb,
                                         const float* __restrict__ Wt,
                                         int bn, int e, float* v) {
  int d = bn & 15, wi = (bn >> 4) & 7, hi = bn >> 7;
  const float* p = pb + (size_t)bn * 16;
  const float* w = Wt + ((size_t)d * 10 + e) * 16;
#pragma unroll
  for (int i = 0; i < 4; ++i)
#pragma unroll
    for (int l = 0; l < 4; ++l) {
      float s = 0.f;
#pragma unroll
      for (int j = 0; j < 4; ++j) s += p[i * 4 + j] * w[j * 4 + l];
      v[i * 4 + l] = s;
    }
  v[0] += (float)hi * 0.125f;
  v[1] += (float)wi * 0.125f;
}

__global__ void __launch_bounds__(320)
class_caps_kernel(const float* __restrict__ pose_in, const float* __restrict__ a_in,
                  const float* __restrict__ Wt, const float* __restrict__ bu,
                  const float* __restrict__ ba, float* __restrict__ out) {
  extern __shared__ float rmat[];  // 1024 * 10 floats (r, reused for ln_ap)
  const int BINS = 1024, E = 10;
  int b = blockIdx.x;
  int e = threadIdx.x >> 5;   // wave id == cout
  int lane = threadIdx.x & 31;
  const float* pb = pose_in + (size_t)b * BINS * 16;
  const float* ab = a_in + (size_t)b * BINS;

  for (int idx = threadIdx.x; idx < BINS * E; idx += blockDim.x) rmat[idx] = 1.f / (float)E;
  __syncthreads();

  float mu[16], sg[16], aout = 0.f;
  for (int it = 0; it < 3; ++it) {
    float rsum = 0.f, musum[16];
#pragma unroll
    for (int c = 0; c < 16; ++c) musum[c] = 0.f;
    for (int bn = lane; bn < BINS; bn += 32) {
      float rw = rmat[bn * E + e] * ab[bn];
      rsum += rw;
      float v[16]; cls_vote(pb, Wt, bn, e, v);
#pragma unroll
      for (int c = 0; c < 16; ++c) musum[c] += rw * v[c];
    }
    rsum = wave_sum32(rsum);
    float inv = 1.f / (rsum + EPSV);
#pragma unroll
    for (int c = 0; c < 16; ++c) mu[c] = wave_sum32(musum[c]) * inv;

    float sgs[16];
#pragma unroll
    for (int c = 0; c < 16; ++c) sgs[c] = 0.f;
    for (int bn = lane; bn < BINS; bn += 32) {
      float rw = rmat[bn * E + e] * ab[bn];
      float v[16]; cls_vote(pb, Wt, bn, e, v);
#pragma unroll
      for (int c = 0; c < 16; ++c) { float d = v[c] - mu[c]; sgs[c] += rw * d * d; }
    }
#pragma unroll
    for (int c = 0; c < 16; ++c) sg[c] = wave_sum32(sgs[c]) * inv + EPSV;

    float cost = 0.f;
#pragma unroll
    for (int c = 0; c < 16; ++c) cost += bu[e] + 0.5f * __logf(sg[c]);
    cost *= rsum;
    aout = sigmoidf_(LAMBDAV * (ba[e] - cost));

    if (it < 2) {
      float lnA = __logf(EPSV + aout);
      __syncthreads();  // all waves done reading r before overwrite with ln_ap
      for (int bn = lane; bn < BINS; bn += 32) {
        float v[16]; cls_vote(pb, Wt, bn, e, v);
        float lp = 0.f;
#pragma unroll
        for (int c = 0; c < 16; ++c) {
          float d = v[c] - mu[c];
          lp += -d * d / (2.f * sg[c]) - 0.5f * __logf(sg[c]) - LN2PI_HALFV;
        }
        rmat[bn * E + e] = lp + lnA;
      }
      __syncthreads();
      for (int bn = threadIdx.x; bn < BINS; bn += blockDim.x) {
        float la[10], mx = -1e30f;
#pragma unroll
        for (int j = 0; j < E; ++j) { la[j] = rmat[bn * E + j]; mx = fmaxf(mx, la[j]); }
        float s = 0.f;
#pragma unroll
        for (int j = 0; j < E; ++j) { la[j] = __expf(la[j] - mx); s += la[j]; }
        float invs = 1.f / s;
#pragma unroll
        for (int j = 0; j < E; ++j) rmat[bn * E + j] = la[j] * invs;
      }
      __syncthreads();
    }
  }
  if (lane == 0) out[(size_t)b * E + e] = aout;
}

// ---------------------------------------------------------------------------
extern "C" void kernel_launch(void* const* d_in, const int* in_sizes, int n_in,
                              void* d_out, int out_size, void* d_ws, size_t ws_size,
                              hipStream_t stream) {
  (void)in_sizes; (void)n_in; (void)out_size; (void)ws_size;
  const float* x     = (const float*)d_in[0];
  const float* c1w   = (const float*)d_in[1];
  const float* c1b   = (const float*)d_in[2];
  const float* ppw   = (const float*)d_in[3];
  const float* ppb   = (const float*)d_in[4];
  const float* paw   = (const float*)d_in[5];
  const float* pab   = (const float*)d_in[6];
  const float* dw1w  = (const float*)d_in[7];
  const float* dw1bu = (const float*)d_in[8];
  const float* dw1ba = (const float*)d_in[9];
  const float* cc1w  = (const float*)d_in[10];
  const float* cc1bu = (const float*)d_in[11];
  const float* cc1ba = (const float*)d_in[12];
  const float* dw2w  = (const float*)d_in[13];
  const float* dw2bu = (const float*)d_in[14];
  const float* dw2ba = (const float*)d_in[15];
  const float* cc2w  = (const float*)d_in[16];
  const float* cc2bu = (const float*)d_in[17];
  const float* cc2ba = (const float*)d_in[18];
  const float* clsw  = (const float*)d_in[19];
  const float* clsbu = (const float*)d_in[20];
  const float* clsba = (const float*)d_in[21];

  float* ws = (float*)d_ws;
  size_t off = 0;
  float* y     = ws + off; off += (size_t)28800 * 64;   // conv1 out
  float* pose1 = ws + off; off += (size_t)28800 * 128;  // primary pose
  float* a1    = ws + off; off += (size_t)28800 * 8;    // primary act
  float* pose2 = ws + off; off += (size_t)7200 * 128;   // dw1 out (32,15,15,128)
  float* a2    = ws + off; off += (size_t)7200 * 8;
  float* pose3 = ws + off; off += (size_t)7200 * 256;   // cc1 out (32,15,15,256)
  float* a3    = ws + off; off += (size_t)7200 * 16;
  float* pose4 = ws + off; off += (size_t)2048 * 256;   // dw2 out (32,8,8,256)
  float* a4    = ws + off; off += (size_t)2048 * 16;
  float* pose5 = ws + off; off += (size_t)2048 * 256;   // cc2 out
  float* a5    = ws + off; off += (size_t)2048 * 16;    // total ~39.3 MB

  conv1_kernel<<<7200, 256, 0, stream>>>(x, c1w, c1b, y);
  prim_pose_wmma<<<1800 * 8, 32, 0, stream>>>(y, ppw, ppb, pose1);
  prim_a_kernel<<<900, 256, 0, stream>>>(y, paw, pab, a1);
  dw_caps_kernel<7><<<3600, 256, 0, stream>>>(pose1, a1, dw1w, dw1bu, dw1ba,
                                              pose2, a2, 32, 30, 30, 8, 15, 15);
  cc_caps_kernel<8><<<3600, 64, 0, stream>>>(pose2, a2, cc1w, cc1bu, cc1ba,
                                             pose3, a3, 7200);
  dw_caps_kernel<5><<<2048, 256, 0, stream>>>(pose3, a3, dw2w, dw2bu, dw2ba,
                                              pose4, a4, 32, 15, 15, 16, 8, 8);
  cc_caps_kernel<16><<<1024, 64, 0, stream>>>(pose4, a4, cc2w, cc2bu, cc2ba,
                                              pose5, a5, 2048);
  class_caps_kernel<<<32, 320, 1024 * 10 * sizeof(float), stream>>>(
      pose5, a5, clsw, clsbu, clsba, (float*)d_out);
}
